// FastMultiheadAttention_82712480186960
// MI455X (gfx1250) — compile-verified, matrix operand order checked by reading, not run on hardware
//
#include <hip/hip_runtime.h>
#include <hip/hip_bf16.h>

// ---------------------------------------------------------------------------
// Fused MHA forward for gfx1250 (MI455X), bf16 WMMA with fp32 accumulation.
// ---------------------------------------------------------------------------

typedef __bf16 bf16_t;
typedef bf16_t v16bf __attribute__((ext_vector_type(16)));
typedef bf16_t v8bf  __attribute__((ext_vector_type(8)));
typedef float  v8f   __attribute__((ext_vector_type(8)));

#define SEQ     2048
#define BATCH   4
#define DIM     1024
#define NHEADS  16
#define HEADDIM 64
#define MROWS   (SEQ * BATCH)     // 8192 rows for the projection GEMMs
#define XSTRIDE (BATCH * DIM)     // 4096: stride between consecutive s in [S,B,D]

// ---------------------------------------------------------------------------
// Fragment helpers. 16-bit A/B matrix layout (CDNA5 ISA 7.12.2):
//   lanes 0-15 : row = lane,    K chunks 0..7  and 16..23
//   lanes 16-31: row = lane-16, K chunks 8..15 and 24..31
// ---------------------------------------------------------------------------
__device__ __forceinline__ v16bf frag_cat(v8bf lo, v8bf hi) {
    v16bf r;
#pragma unroll
    for (int i = 0; i < 8; ++i) { r[i] = lo[i]; r[i + 8] = hi[i]; }
    return r;
}

__device__ __forceinline__ v16bf load_frag_g(const bf16_t* base, int stride, int lane) {
    const bf16_t* p = base + (size_t)(lane & 15) * stride + ((lane >> 4) & 1) * 8;
    v8bf lo = *(const v8bf*)p;
    v8bf hi = *(const v8bf*)(p + 16);
    return frag_cat(lo, hi);
}

__device__ __forceinline__ v8f wmma_bf16(v16bf a, v16bf b, v8f c) {
    return __builtin_amdgcn_wmma_f32_16x16x32_bf16(
        /*neg_a=*/false, a, /*neg_b=*/false, b,
        /*c_mod=*/(short)0, c, /*reuse_a=*/false, /*reuse_b=*/false);
}

// ---------------------------------------------------------------------------
// Fused VOP2+DPP16 butterfly reductions over each 16-lane half (one
// instruction per stage: vD = op(dpp(vS), vS); DPP applies to src0).
// Stage patterns {xor1, xor2, mirror7, mirror15} generate the full 16-lane
// group; later stages' partners differ only in already-uniform bits.
// Per CDNA5 ISA, s_delay_alu is performance-only, so raw asm DPP is safe.
// ---------------------------------------------------------------------------
__device__ __forceinline__ float row_max16(float v) {
    float r;
    asm("v_max_num_f32_dpp %0, %1, %1 quad_perm:[1,0,3,2] row_mask:0xf bank_mask:0xf bound_ctrl:1"
        : "=&v"(r) : "v"(v));
    asm("v_max_num_f32_dpp %0, %1, %1 quad_perm:[2,3,0,1] row_mask:0xf bank_mask:0xf bound_ctrl:1"
        : "=&v"(v) : "v"(r));
    asm("v_max_num_f32_dpp %0, %1, %1 row_half_mirror row_mask:0xf bank_mask:0xf bound_ctrl:1"
        : "=&v"(r) : "v"(v));
    asm("v_max_num_f32_dpp %0, %1, %1 row_mirror row_mask:0xf bank_mask:0xf bound_ctrl:1"
        : "=&v"(v) : "v"(r));
    return v;
}

__device__ __forceinline__ float row_sum16(float v) {
    float r;
    asm("v_add_f32_dpp %0, %1, %1 quad_perm:[1,0,3,2] row_mask:0xf bank_mask:0xf bound_ctrl:1"
        : "=&v"(r) : "v"(v));
    asm("v_add_f32_dpp %0, %1, %1 quad_perm:[2,3,0,1] row_mask:0xf bank_mask:0xf bound_ctrl:1"
        : "=&v"(v) : "v"(r));
    asm("v_add_f32_dpp %0, %1, %1 row_half_mirror row_mask:0xf bank_mask:0xf bound_ctrl:1"
        : "=&v"(r) : "v"(v));
    asm("v_add_f32_dpp %0, %1, %1 row_mirror row_mask:0xf bank_mask:0xf bound_ctrl:1"
        : "=&v"(v) : "v"(r));
    return v;
}

// ---------------------------------------------------------------------------
// fp32 -> bf16 convert
// ---------------------------------------------------------------------------
__global__ void f32_to_bf16(const float* __restrict__ in, bf16_t* __restrict__ out, int n) {
    int i = blockIdx.x * 256 + threadIdx.x;
    if (i < n) out[i] = (bf16_t)in[i];
}

// ---------------------------------------------------------------------------
// GEMM: Y[M=8192, N=1024] = A[8192,1024] * W^T, K=1024.
// Block: 256 threads = 8 waves (2x4), wave tile 32x64, block tile 64x256.
// MODE 0: fp32 out, row-major [M,N]           (final projection)
// MODE 1: bf16 out, row-major [M,N]           (Q, K: consumed in x-layout)
// MODE 2: bf16 out scattered to [b,h,d,t]     (V transposed)
// ---------------------------------------------------------------------------
template <int MODE>
__global__ __launch_bounds__(256) void gemm_qkvo(const bf16_t* __restrict__ A,
                                                 const bf16_t* __restrict__ W,
                                                 void* __restrict__ dst) {
    const int lane  = threadIdx.x & 31;
    const int wave  = threadIdx.x >> 5;
    const int mBase = blockIdx.x * 64 + (wave >> 2) * 32;
    const int nBase = blockIdx.y * 256 + (wave & 3) * 64;

    v8f acc[2][4];
#pragma unroll
    for (int mi = 0; mi < 2; ++mi)
#pragma unroll
        for (int ni = 0; ni < 4; ++ni)
#pragma unroll
            for (int r = 0; r < 8; ++r) acc[mi][ni][r] = 0.0f;

    const bf16_t* Ab = A + (size_t)mBase * DIM;
    const bf16_t* Wb = W + (size_t)nBase * DIM;

    for (int k = 0; k < DIM; k += 32) {
        // Prefetch two K-steps ahead (-> global_prefetch_b8).
        if (k + 64 < DIM) {
            __builtin_prefetch(Ab + (size_t)(lane & 15) * DIM + k + 64, 0, 3);
            __builtin_prefetch(Wb + (size_t)(lane & 15) * DIM + k + 64, 0, 3);
        }
        v16bf a0 = load_frag_g(Ab + k, DIM, lane);
        v16bf a1 = load_frag_g(Ab + (size_t)16 * DIM + k, DIM, lane);
        v16bf b[4];
#pragma unroll
        for (int ni = 0; ni < 4; ++ni)
            b[ni] = load_frag_g(Wb + (size_t)(ni * 16) * DIM + k, DIM, lane);
#pragma unroll
        for (int ni = 0; ni < 4; ++ni) {
            acc[0][ni] = wmma_bf16(a0, b[ni], acc[0][ni]);
            acc[1][ni] = wmma_bf16(a1, b[ni], acc[1][ni]);
        }
    }

    // C/D layout: VGPR r, lanes 0-15 -> M=r, N=lane; lanes 16-31 -> M=8+r
    const int rowAdd = ((lane >> 4) & 1) * 8;
    const int col    = lane & 15;

    if constexpr (MODE == 0) {
        float* po = (float*)dst + (size_t)(mBase + rowAdd) * DIM + nBase + col;
#pragma unroll
        for (int mi = 0; mi < 2; ++mi)
#pragma unroll
            for (int ni = 0; ni < 4; ++ni)
#pragma unroll
                for (int r = 0; r < 8; ++r)
                    po[(size_t)(mi * 16 + r) * DIM + ni * 16] = acc[mi][ni][r];
    } else if constexpr (MODE == 1) {
        bf16_t* po = (bf16_t*)dst + (size_t)(mBase + rowAdd) * DIM + nBase + col;
#pragma unroll
        for (int mi = 0; mi < 2; ++mi)
#pragma unroll
            for (int ni = 0; ni < 4; ++ni)
#pragma unroll
                for (int r = 0; r < 8; ++r)
                    po[(size_t)(mi * 16 + r) * DIM + ni * 16] = (bf16_t)acc[mi][ni][r];
    } else {
        // V^T scatter: i = ibase + r, ibase % 8 == 0 => b = r&3, t = ibase/4 + r/4.
        const int ibase = mBase + rowAdd;
        const int tb    = ibase >> 2;
#pragma unroll
        for (int ni = 0; ni < 4; ++ni) {
            int j = nBase + ni * 16 + col;
            int h = j >> 6, d = j & 63;
            bf16_t* po = (bf16_t*)dst + ((size_t)h * HEADDIM + d) * SEQ + tb;
#pragma unroll
            for (int mi = 0; mi < 2; ++mi)
#pragma unroll
                for (int r = 0; r < 8; ++r) {
                    int bb = r & 3;
                    po[(size_t)bb * (NHEADS * HEADDIM * SEQ) + mi * 4 + (r >> 2)] =
                        (bf16_t)acc[mi][ni][r];
                }
        }
    }
}

// ---------------------------------------------------------------------------
// Flash attention: one wave per (b,h, 16-query tile). Online softmax over
// 2048 keys in steps of 32. Q/K are read directly from x-layout [S,B,D]
// (row stride 4096 elems, contiguous in d). V^T from [b,h,d,t].
// P tile converts C/D layout -> A layout via a 1 KB LDS bounce.
// Row reductions are single-instruction fused DPP16 VALU butterflies.
// ---------------------------------------------------------------------------
__global__ __launch_bounds__(32) void attn_fwd(const bf16_t* __restrict__ qb,
                                               const bf16_t* __restrict__ kb,
                                               const bf16_t* __restrict__ vtb,
                                               const float* __restrict__ mask,
                                               bf16_t* __restrict__ ctxb) {
    __shared__ alignas(16) bf16_t pbuf[16 * 32];   // 1 KB, one wave per WG

    const int lane  = threadIdx.x;
    const int bh    = blockIdx.y;                   // b*16 + h
    const int bb    = bh >> 4;
    const int h     = bh & 15;
    const int sBase = blockIdx.x * 16;

    // Q/K live in x-layout [S,B,D]: row s of head h at ((s*B+b)*D + h*64).
    const bf16_t* q  = qb  + ((size_t)sBase * BATCH + bb) * DIM + h * HEADDIM;
    const bf16_t* kk = kb  + (size_t)bb * DIM + h * HEADDIM;
    const bf16_t* vt = vtb + (size_t)bh * HEADDIM * SEQ;   // [d][t]

    // Q tile A-fragments (16 rows x K=64 -> two K=32 fragments), loaded once.
    v16bf qa0 = load_frag_g(q, XSTRIDE, lane);
    v16bf qa1 = load_frag_g(q + 32, XSTRIDE, lane);

    const int col    = lane & 15;
    const int rowAdd = ((lane >> 4) & 1) * 8;

    float m[8], l[8];
    v8f   c[4];
#pragma unroll
    for (int r = 0; r < 8; ++r) { m[r] = -1e30f; l[r] = 0.0f; }
#pragma unroll
    for (int nd = 0; nd < 4; ++nd)
#pragma unroll
        for (int r = 0; r < 8; ++r) c[nd][r] = 0.0f;

    const float* maskp = mask + (size_t)(sBase + rowAdd) * SEQ + col;

    for (int t = 0; t < SEQ; t += 32) {
        // K B-fragments: column n of B = key row t+n (x-layout, stride 4096).
        const bf16_t* kt = kk + (size_t)t * XSTRIDE;
        v16bf k00 = load_frag_g(kt, XSTRIDE, lane);                          // t..t+15, d 0..31
        v16bf k01 = load_frag_g(kt + 32, XSTRIDE, lane);                     // d 32..63
        v16bf k10 = load_frag_g(kt + (size_t)16 * XSTRIDE, XSTRIDE, lane);
        v16bf k11 = load_frag_g(kt + (size_t)16 * XSTRIDE + 32, XSTRIDE, lane);

        v8f s0, s1;
#pragma unroll
        for (int r = 0; r < 8; ++r) { s0[r] = 0.0f; s1[r] = 0.0f; }
        s0 = wmma_bf16(qa0, k00, s0);
        s0 = wmma_bf16(qa1, k01, s0);
        s1 = wmma_bf16(qa0, k10, s1);
        s1 = wmma_bf16(qa1, k11, s1);

        // Online softmax: rows of this lane's half are (sBase+rowAdd+r).
        float alpha[8];
#pragma unroll
        for (int r = 0; r < 8; ++r) {
            float v0 = s0[r] * 0.125f + maskp[(size_t)r * SEQ + t];
            float v1 = s1[r] * 0.125f + maskp[(size_t)r * SEQ + t + 16];
            float mx = row_max16(fmaxf(v0, v1));
            float mn = fmaxf(m[r], mx);
            float a  = __expf(m[r] - mn);
            float p0 = __expf(v0 - mn);
            float p1 = __expf(v1 - mn);
            float rs = row_sum16(p0 + p1);
            l[r]     = l[r] * a + rs;
            m[r]     = mn;
            alpha[r] = a;
            int prow = rowAdd + r;
            pbuf[prow * 32 + col]      = (bf16_t)p0;
            pbuf[prow * 32 + 16 + col] = (bf16_t)p1;
        }

        // Rescale running context by alpha (per-row).
#pragma unroll
        for (int nd = 0; nd < 4; ++nd)
#pragma unroll
            for (int r = 0; r < 8; ++r) c[nd][r] *= alpha[r];

        // Single-wave workgroup: only need DS-counter drain before reload.
        asm volatile("s_wait_dscnt 0" ::: "memory");

        // Reload P as A-fragment from row-major LDS tile (stride 32 elems).
        {
            const bf16_t* pp = &pbuf[(lane & 15) * 32 + ((lane >> 4) & 1) * 8];
            v8bf lo = *(const v8bf*)pp;
            v8bf hi = *(const v8bf*)(pp + 16);
            v16bf pa = frag_cat(lo, hi);

            // V^T B-fragments: column n = d, stored [d][t] -> contiguous in t.
#pragma unroll
            for (int nd = 0; nd < 4; ++nd) {
                v16bf vb = load_frag_g(vt + (size_t)(nd * 16) * SEQ + t, SEQ, lane);
                c[nd] = wmma_bf16(pa, vb, c[nd]);
            }
        }
    }

    // Epilogue: divide by row sums, store to ctx[S,B,D] (D index = h*64+d).
    bf16_t* po = ctxb + (((size_t)(sBase + rowAdd) * BATCH + bb)) * DIM + h * HEADDIM + col;
#pragma unroll
    for (int nd = 0; nd < 4; ++nd)
#pragma unroll
        for (int r = 0; r < 8; ++r) {
            float v = c[nd][r] / l[r];
            po[(size_t)r * XSTRIDE + nd * 16] = (bf16_t)v;
        }
}

// ---------------------------------------------------------------------------
// Launch: convert -> 3x projection GEMM -> attention -> output GEMM.
// ---------------------------------------------------------------------------
extern "C" void kernel_launch(void* const* d_in, const int* in_sizes, int n_in,
                              void* d_out, int out_size, void* d_ws, size_t ws_size,
                              hipStream_t stream) {
    (void)in_sizes; (void)n_in; (void)out_size; (void)ws_size;

    const float* x    = (const float*)d_in[0];
    const float* mask = (const float*)d_in[1];
    const float* wq   = (const float*)d_in[2];
    const float* wk   = (const float*)d_in[3];
    const float* wv   = (const float*)d_in[4];
    const float* wo   = (const float*)d_in[5];

    char* ws = (char*)d_ws;
    size_t off = 0;
    bf16_t* xb   = (bf16_t*)(ws + off); off += (size_t)MROWS * DIM * 2;   // 16 MB
    bf16_t* wqb  = (bf16_t*)(ws + off); off += (size_t)DIM * DIM * 2;     //  2 MB
    bf16_t* wkb  = (bf16_t*)(ws + off); off += (size_t)DIM * DIM * 2;
    bf16_t* wvb  = (bf16_t*)(ws + off); off += (size_t)DIM * DIM * 2;
    bf16_t* wob  = (bf16_t*)(ws + off); off += (size_t)DIM * DIM * 2;
    bf16_t* qbuf = (bf16_t*)(ws + off); off += (size_t)MROWS * DIM * 2;   // [S,B,D] bf16
    bf16_t* kbuf = (bf16_t*)(ws + off); off += (size_t)MROWS * DIM * 2;   // [S,B,D] bf16
    bf16_t* vtb  = (bf16_t*)(ws + off); off += (size_t)MROWS * DIM * 2;   // [b,h,d,t]
    bf16_t* ctxb = (bf16_t*)(ws + off); off += (size_t)MROWS * DIM * 2;   // [S,B,D] bf16

    const int nx = MROWS * DIM;
    const int nw = DIM * DIM;
    f32_to_bf16<<<(nx + 255) / 256, 256, 0, stream>>>(x,  xb,  nx);
    f32_to_bf16<<<(nw + 255) / 256, 256, 0, stream>>>(wq, wqb, nw);
    f32_to_bf16<<<(nw + 255) / 256, 256, 0, stream>>>(wk, wkb, nw);
    f32_to_bf16<<<(nw + 255) / 256, 256, 0, stream>>>(wv, wvb, nw);
    f32_to_bf16<<<(nw + 255) / 256, 256, 0, stream>>>(wo, wob, nw);

    dim3 gg(MROWS / 64, DIM / 256);   // 128 x 4
    gemm_qkvo<1><<<gg, 256, 0, stream>>>(xb, wqb, qbuf);
    gemm_qkvo<1><<<gg, 256, 0, stream>>>(xb, wkb, kbuf);
    gemm_qkvo<2><<<gg, 256, 0, stream>>>(xb, wvb, vtb);

    attn_fwd<<<dim3(SEQ / 16, BATCH * NHEADS), 32, 0, stream>>>(qbuf, kbuf, vtb, mask, ctxb);

    gemm_qkvo<0><<<gg, 256, 0, stream>>>(ctxb, wob, d_out);
}
